// MGRUCell_67070209295131
// MI455X (gfx1250) — compile-verified, matrix operand order checked by reading, not run on hardware
//
#include <hip/hip_runtime.h>

// ---------------------------------------------------------------------------
// MGRU cell for MI455X (gfx1250): bf16 WMMA GEMMs + fused epilogues.
// Tile staging uses GLOBAL_LOAD_ASYNC_TO_LDS_B128 (ASYNCcnt-tracked).
// Shapes: B=4096, I=H=O=512, K(slices)=16.
// ---------------------------------------------------------------------------

typedef __attribute__((ext_vector_type(16))) __bf16        v16bf;
typedef __attribute__((ext_vector_type(8)))  float         v8f;
typedef __attribute__((ext_vector_type(4))) unsigned int   uint4v;

union Frag { v16bf v; uint4v u[2]; };

#define BATCH 4096
#define HID   512
#define NSLC  16
#define BH    (BATCH * HID)

__device__ __forceinline__ unsigned short f2bf(float f) {
    unsigned int u = __float_as_uint(f);
    unsigned int r = (u + 0x7fffu + ((u >> 16) & 1u)) >> 16;
    return (unsigned short)r;
}

__device__ __forceinline__ float sigmoidf(float x) {
    return 1.0f / (1.0f + __expf(-x));
}

// ------------------------------ packing kernels ----------------------------

// Wcat (K=1536 x N=1536), stored transposed [n][k] bf16:
//   k<1024 : n<512 -> Wu[k][n] ; n<1024 -> Wr[k][n-512] ; else Wm[k][n-1024]
//   k>=1024: n<1024 -> 0 ; else Wm[k][n-1024]
__global__ void k_pack_wcat(unsigned short* __restrict__ dst,
                            const float* __restrict__ Wu,
                            const float* __restrict__ Wr,
                            const float* __restrict__ Wm) {
    int idx = blockIdx.x * 256 + threadIdx.x;        // over 1536*1536
    int n = idx / 1536;
    int k = idx - n * 1536;
    float v;
    if (k < 1024) {
        if      (n < 512)  v = Wu[k * 512 + n];
        else if (n < 1024) v = Wr[k * 512 + (n - 512)];
        else               v = Wm[k * 512 + (n - 1024)];
    } else {
        v = (n < 1024) ? 0.0f : Wm[k * 512 + (n - 1024)];
    }
    dst[idx] = f2bf(v);
}

// Generic transpose+convert: src [Kd x Nd] f32 row-major -> dst [Nd x Kd] bf16
__global__ void k_transpose_bf16(unsigned short* __restrict__ dst,
                                 const float* __restrict__ src,
                                 int Kd, int Nd) {
    int idx = blockIdx.x * 256 + threadIdx.x;        // over Kd*Nd
    int n = idx / Kd;
    int k = idx - n * Kd;
    dst[idx] = f2bf(src[k * Nd + n]);
}

// A_gates [B x 1536] bf16 = [sample | hiddens[15] | mem_para]
__global__ void k_build_Agates(unsigned short* __restrict__ dst,
                               const float* __restrict__ sample,
                               const float* __restrict__ hiddens,
                               const float* __restrict__ mem_para) {
    int idx = blockIdx.x * 256 + threadIdx.x;        // over B*1536
    int b = idx / 1536;
    int j = idx - b * 1536;
    float v;
    if      (j < 512)  v = sample[b * 512 + j];
    else if (j < 1024) v = hiddens[(size_t)15 * BH + b * 512 + (j - 512)];
    else               v = mem_para[b * 512 + (j - 1024)];
    dst[idx] = f2bf(v);
}

// A_h [B x 1024] bf16 = [sample | R * hidden]
__global__ void k_build_Ah(unsigned short* __restrict__ dst,
                           const float* __restrict__ sample,
                           const float* __restrict__ R,
                           const float* __restrict__ hiddens) {
    int idx = blockIdx.x * 256 + threadIdx.x;        // over B*1024
    int b = idx >> 10;
    int j = idx & 1023;
    float v;
    if (j < 512) v = sample[b * 512 + j];
    else {
        int h = j - 512;
        v = R[b * 512 + h] * hiddens[(size_t)15 * BH + b * 512 + h];
    }
    dst[idx] = f2bf(v);
}

// ------------------------------- GEMM core ---------------------------------
// C[M x N] f32 = A[M x K](bf16, row-major) * Bt[N x K](bf16, row-major-of-B^T)
// Block tile 128x128, 8 waves (4 M x 2 N), wave tile 32x64, BK = 64
// (two WMMA K-steps per LDS stage). Double-buffered LDS (64 KB of 320 KB),
// filled with GLOBAL_LOAD_ASYNC_TO_LDS_B128; v_wmma_f32_16x16x32_bf16.

// One 128x64-bf16 tile (1024 x 16B) per workgroup: each thread issues 4
// async b128 transfers straight into LDS (no VGPR staging).
__device__ __forceinline__ void tile_async_load(const unsigned short* __restrict__ base,
                                                int ld, int r0, int kt, int tid,
                                                uint4v* lsbuf) {
#pragma unroll
    for (int it = 0; it < 4; ++it) {
        int e8  = tid + it * 256;            // 1024 x uint4v = 128 rows x 64 cols
        int row = e8 >> 3;
        int c8  = e8 & 7;
        const void* gp = base + (size_t)(r0 + row) * ld + (size_t)kt * 64 + c8 * 8;
        unsigned lds = (unsigned)(uintptr_t)(lsbuf + e8);  // LDS aperture: offset in [31:0]
        asm volatile("global_load_async_to_lds_b128 %0, %1, off"
                     :: "v"(lds), "v"(gp) : "memory");
    }
}

__device__ __forceinline__ void wait_async0() {
    asm volatile("s_wait_asynccnt 0x0" ::: "memory");
}

template <typename Epi>
__device__ __forceinline__ void gemm_core(const unsigned short* __restrict__ A,
                                          const unsigned short* __restrict__ Bt,
                                          int Kdim, Epi&& epi) {
    constexpr int BM = 128, BN = 128;
    __shared__ uint4v lsA[2][BM * 8];        // [buf][row*8 + c8], 16 KB each
    __shared__ uint4v lsB[2][BN * 8];

    const int tid  = threadIdx.x;
    const int lane = tid & 31;
    const int wave = tid >> 5;
    const int wm   = wave >> 1;              // 0..3 -> M offset wm*32
    const int wn   = wave & 1;               // 0..1 -> N offset wn*64
    const int bm0  = blockIdx.y * BM;
    const int bn0  = blockIdx.x * BN;

    const int ln15 = lane & 15;
    const int kb8  = (lane < 16) ? 0 : 1;    // fragment K-subblock (8-elt units)

    v8f acc[2][4] = {};

    // prologue: stage 0 -> buffer 0 (async memory -> LDS)
    tile_async_load(A,  Kdim, bm0, 0, tid, lsA[0]);
    tile_async_load(Bt, Kdim, bn0, 0, tid, lsB[0]);
    wait_async0();
    __syncthreads();

    const int nk = Kdim >> 6;                // stages of BK=64
    int cur = 0;
    for (int kt = 0; kt < nk; ++kt) {
        const bool pre = (kt + 1 < nk);
        if (pre) {                           // fill next buffer while computing
            int nxt = cur ^ 1;
            tile_async_load(A,  Kdim, bm0, kt + 1, tid, lsA[nxt]);
            tile_async_load(Bt, Kdim, bn0, kt + 1, tid, lsB[nxt]);
        }
        if (kt + 2 < nk) {   // pull the stage after next into L2 (global_prefetch_b8)
            __builtin_prefetch(A  + (size_t)(bm0 + (tid >> 1)) * Kdim + (size_t)(kt + 2) * 64, 0, 1);
            __builtin_prefetch(Bt + (size_t)(bn0 + (tid >> 1)) * Kdim + (size_t)(kt + 2) * 64, 0, 1);
        }

        // two WMMA K-steps per stage; 6 fragments live at a time
#pragma unroll
        for (int ks = 0; ks < 2; ++ks) {
            const int ks4 = ks * 4;
            Frag fa[2], fb[4];
#pragma unroll
            for (int mt = 0; mt < 2; ++mt) {
                int row = wm * 32 + mt * 16 + ln15;
                fa[mt].u[0] = lsA[cur][row * 8 + ks4 + kb8];
                fa[mt].u[1] = lsA[cur][row * 8 + ks4 + kb8 + 2];
            }
#pragma unroll
            for (int nt = 0; nt < 4; ++nt) {
                int row = wn * 64 + nt * 16 + ln15;
                fb[nt].u[0] = lsB[cur][row * 8 + ks4 + kb8];
                fb[nt].u[1] = lsB[cur][row * 8 + ks4 + kb8 + 2];
            }
#pragma unroll
            for (int mt = 0; mt < 2; ++mt)
#pragma unroll
                for (int nt = 0; nt < 4; ++nt)
                    acc[mt][nt] = __builtin_amdgcn_wmma_f32_16x16x32_bf16(
                        false, fa[mt].v, false, fb[nt].v,
                        (short)0, acc[mt][nt], false, false);
        }

        wait_async0();                       // own async writes to LDS done
        __syncthreads();                     // everyone's writes visible
        cur ^= 1;
    }

    // epilogue (ISA C/D layout: row = r + 8*(lane>=16), col = lane&15)
    const int rsel = (lane >> 4) << 3;
#pragma unroll
    for (int mt = 0; mt < 2; ++mt)
#pragma unroll
        for (int nt = 0; nt < 4; ++nt)
#pragma unroll
            for (int r = 0; r < 8; ++r) {
                int gm = bm0 + wm * 32 + mt * 16 + rsel + r;
                int gn = bn0 + wn * 64 + nt * 16 + ln15;
                epi(gm, gn, acc[mt][nt][r]);
            }
}

// ------------------------------ GEMM kernels -------------------------------

__global__ void __launch_bounds__(256)
k_gemm_gates(const unsigned short* __restrict__ Ag,
             const unsigned short* __restrict__ Wcat_t,
             const float* __restrict__ bu,
             const float* __restrict__ br,
             const float* __restrict__ bm,
             float* __restrict__ U,
             float* __restrict__ R,
             float* __restrict__ MG) {
    gemm_core(Ag, Wcat_t, 1536, [&](int gm, int gn, float x) {
        if (gn < 512)       U[(size_t)gm * 512 + gn]          = sigmoidf(x + bu[gn]);
        else if (gn < 1024) R[(size_t)gm * 512 + (gn - 512)]  = sigmoidf(x + br[gn - 512]);
        else                MG[(size_t)gm * 512 + (gn - 1024)] =
                                0.5f * sigmoidf(x + bm[gn - 1024]);
    });
}

__global__ void __launch_bounds__(256)
k_gemm_htilde(const unsigned short* __restrict__ Ah,
              const unsigned short* __restrict__ Wh_t,
              const float* __restrict__ bh,
              const float* __restrict__ U,
              const float* __restrict__ MG,
              const float* __restrict__ hiddens,
              float* __restrict__ uh_out,       // d_out + B*O
              unsigned short* __restrict__ hnbf) {
    gemm_core(Ah, Wh_t, 1024, [&](int gm, int gn, float x) {
        float ht = tanhf(x + bh[gn]);
        size_t e = (size_t)gm * 512 + gn;
        float d  = MG[e];
        // w[15-i] = prod_{m=0..i} (m-d)/(m+1); filtered = sum_k hiddens[k]*w[k]
        float p = 1.0f, filt = 0.0f;
        const float* hp = hiddens + e;
#pragma unroll
        for (int i = 0; i < 16; ++i) {
            p    *= ((float)i - d) * (1.0f / (float)(i + 1));
            filt += p * hp[(size_t)(15 - i) * BH];
        }
        float hn = -filt + U[e] * ht;
        uh_out[(size_t)15 * BH + e] = hn;     // updated_hiddens slice 15
        hnbf[e] = f2bf(hn);
    });
}

__global__ void __launch_bounds__(256)
k_gemm_out(const unsigned short* __restrict__ hnbf,
           const unsigned short* __restrict__ Wo_t,
           const float* __restrict__ bo,
           float* __restrict__ out) {
    gemm_core(hnbf, Wo_t, 512, [&](int gm, int gn, float x) {
        out[(size_t)gm * 512 + gn] = x + bo[gn];
    });
}

// ------------------------------- launcher ----------------------------------

extern "C" void kernel_launch(void* const* d_in, const int* in_sizes, int n_in,
                              void* d_out, int out_size, void* d_ws, size_t ws_size,
                              hipStream_t stream) {
    const float* sample   = (const float*)d_in[0];
    const float* hiddens  = (const float*)d_in[1];
    const float* mem_para = (const float*)d_in[2];
    const float* Wu = (const float*)d_in[3];
    const float* bu = (const float*)d_in[4];
    const float* Wr = (const float*)d_in[5];
    const float* br = (const float*)d_in[6];
    const float* Wm = (const float*)d_in[7];
    const float* bm = (const float*)d_in[8];
    const float* Wh = (const float*)d_in[9];
    const float* bh = (const float*)d_in[10];
    const float* Wo = (const float*)d_in[11];
    const float* bo = (const float*)d_in[12];

    float* out   = (float*)d_out;                      // [B, O]
    float* uh    = out + (size_t)BATCH * 512;          // [16, B, H]
    float* memgt = uh + (size_t)NSLC * BH;             // [B, H]

    // workspace carve (256B aligned regions)
    char* ws = (char*)d_ws;
    size_t off = 0;
    auto carve = [&](size_t bytes) {
        char* p = ws + off;
        off = (off + bytes + 255) & ~(size_t)255;
        return p;
    };
    unsigned short* wcat_t = (unsigned short*)carve((size_t)1536 * 1536 * 2);
    unsigned short* wh_t   = (unsigned short*)carve((size_t)512 * 1024 * 2);
    unsigned short* wo_t   = (unsigned short*)carve((size_t)512 * 512 * 2);
    unsigned short* Ag     = (unsigned short*)carve((size_t)BATCH * 1536 * 2);
    unsigned short* Ah     = (unsigned short*)carve((size_t)BATCH * 1024 * 2);
    unsigned short* hnbf   = (unsigned short*)carve((size_t)BH * 2);
    float*          U      = (float*)carve((size_t)BH * 4);
    float*          R      = (float*)carve((size_t)BH * 4);
    (void)ws_size; (void)in_sizes; (void)n_in; (void)out_size;

    // 1) pack weights (bf16, transposed [N][K])
    k_pack_wcat     <<<(1536 * 1536) / 256, 256, 0, stream>>>(wcat_t, Wu, Wr, Wm);
    k_transpose_bf16<<<(1024 * 512) / 256, 256, 0, stream>>>(wh_t, Wh, 1024, 512);
    k_transpose_bf16<<<(512 * 512)  / 256, 256, 0, stream>>>(wo_t, Wo, 512, 512);

    // 2) shift hiddens: updated_hiddens[0:15] = hiddens[1:16]
    hipMemcpyAsync(uh, hiddens + BH, (size_t)(NSLC - 1) * BH * sizeof(float),
                   hipMemcpyDeviceToDevice, stream);

    // 3) gates GEMM: [sample|hidden|mem_para] @ [Wu|Wr|Wm]
    k_build_Agates<<<((size_t)BATCH * 1536) / 256, 256, 0, stream>>>(
        Ag, sample, hiddens, mem_para);
    k_gemm_gates<<<dim3(1536 / 128, BATCH / 128), 256, 0, stream>>>(
        Ag, wcat_t, bu, br, bm, U, R, memgt);

    // 4) h_tilde GEMM + fused mem_filter -> hidden_now
    k_build_Ah<<<((size_t)BATCH * 1024) / 256, 256, 0, stream>>>(
        Ah, sample, R, hiddens);
    k_gemm_htilde<<<dim3(512 / 128, BATCH / 128), 256, 0, stream>>>(
        Ah, wh_t, bh, U, memgt, hiddens, uh, hnbf);

    // 5) output GEMM
    k_gemm_out<<<dim3(512 / 128, BATCH / 128), 256, 0, stream>>>(
        hnbf, wo_t, bo, out);
}
